// HyenaBlock_27522150433487
// MI455X (gfx1250) — compile-verified, matrix operand order checked by reading, not run on hardware
//
#include <hip/hip_runtime.h>
#include <hip/hip_bf16.h>
#include <math.h>

// ---------- sizes ----------
#define LEN     4096
#define DIM     1024
#define HID     2048          // 2*DIM
#define DIMPOS  256
#define BATCH   8
#define NFFT    8192          // 2*LEN
#define NROWS   (BATCH * LEN) // 32768 token rows

#define MTILE   32            // token rows per block in the MLP kernel
#define XS_LD   (DIM + 8)     // padded LDS row strides (bank-conflict free:
#define YS_LD   (HID + 8)     //  stride % 256B == 16B)

// ---------- vector types ----------
typedef __attribute__((ext_vector_type(16))) __bf16   v16bf;
typedef __attribute__((ext_vector_type(8)))  __bf16   v8bf;
typedef __attribute__((ext_vector_type(8)))  float    v8f;
typedef __attribute__((ext_vector_type(4)))  float    f4;
typedef __attribute__((ext_vector_type(4)))  unsigned int u4;

// =====================================================================
// Kernel 0: f32 -> bf16 pre-conversion (x, W1, W2). Weights become 8 MB
// total => permanently L2-resident (192 MB L2).
// =====================================================================
__global__ __launch_bounds__(256) void f32_to_bf16_kernel(const float* __restrict__ src,
                                                          __bf16* __restrict__ dst,
                                                          long n) {
    long i = ((long)blockIdx.x * 256 + threadIdx.x) * 8;
    if (i + 8 > n) return;
    f4 a = *(const f4*)(src + i);
    f4 b = *(const f4*)(src + i + 4);
    v8bf r;
    #pragma unroll
    for (int e = 0; e < 4; ++e) {
        r[e]     = (__bf16)a[e];
        r[4 + e] = (__bf16)b[e];
    }
    *(v8bf*)(dst + i) = r;
}

// =====================================================================
// Kernel 1: LayerNorm
// =====================================================================
__global__ __launch_bounds__(256) void ln_kernel(const float* __restrict__ z,
                                                 const float* __restrict__ gamma,
                                                 const float* __restrict__ beta,
                                                 float* __restrict__ zn) {
    const int row = blockIdx.x;
    const float* zr = z + (size_t)row * DIM;
    float s = 0.f, ss = 0.f;
    for (int i = threadIdx.x; i < DIM; i += 256) {
        float v = zr[i];
        s += v; ss += v * v;
    }
    for (int o = 16; o > 0; o >>= 1) {
        s  += __shfl_xor(s,  o, 32);
        ss += __shfl_xor(ss, o, 32);
    }
    __shared__ float ps[8], pss[8];
    const int wid = threadIdx.x >> 5, lid = threadIdx.x & 31;
    if (lid == 0) { ps[wid] = s; pss[wid] = ss; }
    __syncthreads();
    if (wid == 0) {
        s  = (lid < 8) ? ps[lid]  : 0.f;
        ss = (lid < 8) ? pss[lid] : 0.f;
        for (int o = 4; o > 0; o >>= 1) {
            s  += __shfl_xor(s,  o, 32);
            ss += __shfl_xor(ss, o, 32);
        }
        if (lid == 0) { ps[0] = s * (1.f / DIM); pss[0] = ss * (1.f / DIM); }
    }
    __syncthreads();
    const float mu   = ps[0];
    const float var  = pss[0] - mu * mu;
    const float rstd = rsqrtf(var + 1e-5f);
    float* o = zn + (size_t)row * DIM;
    for (int i = threadIdx.x; i < DIM; i += 256)
        o[i] = (zr[i] - mu) * rstd * gamma[i] + beta[i];
}

// =====================================================================
// 8192-point radix-2 Stockham complex FFT in LDS (self-sorting).
// =====================================================================
__device__ float2* fft8192(float2* X, float2* Y, int tid, int nthr, float dirSign) {
    int l = NFFT / 2;   // 4096
    int mshift = 0;
    while (l >= 1) {
        const int m = 1 << mshift;
        __syncthreads();
        for (int i = tid; i < NFFT / 2; i += nthr) {
            const int k = i & (m - 1);
            const int j = i >> mshift;
            float2 c0 = X[i];
            float2 c1 = X[i + NFFT / 2];
            float2 d; d.x = c0.x - c1.x; d.y = c0.y - c1.y;
            const float th = dirSign * (-3.14159265358979323846f) * (float)j / (float)l;
            float sn, cs;
            __sincosf(th, &sn, &cs);
            float2 e; e.x = c0.x + c1.x; e.y = c0.y + c1.y;
            float2 w; w.x = cs * d.x - sn * d.y; w.y = cs * d.y + sn * d.x;
            Y[k + 2 * j * m]     = e;
            Y[k + 2 * j * m + m] = w;
        }
        float2* t = X; X = Y; Y = t;
        l >>= 1; ++mshift;
    }
    __syncthreads();
    return X;
}

// =====================================================================
// Kernel 2: filter spectra, TWO channels per FFT (re/im packing).
// =====================================================================
__global__ __launch_bounds__(256) void filter_fft_kernel(const float* __restrict__ a,
                                                         const float* __restrict__ pe,
                                                         const float* __restrict__ W_pos,
                                                         float2* __restrict__ Wf) {
    const int d0 = blockIdx.x * 2;
    __shared__ float wrow0[DIMPOS], wrow1[DIMPOS];
    __shared__ float2 buf0[NFFT];
    __shared__ float2 buf1[NFFT];
    for (int i = threadIdx.x; i < DIMPOS; i += 256) {
        wrow0[i] = W_pos[(size_t)d0 * DIMPOS + i];
        wrow1[i] = W_pos[(size_t)(d0 + 1) * DIMPOS + i];
    }
    __syncthreads();
    const float decay = __expf(a[0]);
    for (int l = threadIdx.x; l < LEN; l += 256) {
        const float* per = pe + (size_t)l * DIMPOS;
        float a0 = 0.f, a1 = 0.f;
        #pragma unroll 8
        for (int k = 0; k < DIMPOS; ++k) {
            float p = per[k];
            a0 += p * wrow0[k];
            a1 += p * wrow1[k];
        }
        const float win = __expf(-(float)l * decay);
        float2 v; v.x = a0 * win; v.y = a1 * win;
        buf0[l] = v;
    }
    for (int l = LEN + threadIdx.x; l < NFFT; l += 256) {
        float2 zv; zv.x = 0.f; zv.y = 0.f;
        buf0[l] = zv;
    }
    float2* Z = fft8192(buf0, buf1, threadIdx.x, 256, 1.0f);
    for (int k = threadIdx.x; k < NFFT; k += 256) {
        const int nk = (NFFT - k) & (NFFT - 1);
        float2 zk = Z[k], zn_ = Z[nk];
        float2 Ha; Ha.x = 0.5f * (zk.x + zn_.x); Ha.y = 0.5f * (zk.y - zn_.y);
        float2 Hb; Hb.x = 0.5f * (zk.y + zn_.y); Hb.y = -0.5f * (zk.x - zn_.x);
        Wf[(size_t)d0 * NFFT + k]       = Ha;
        Wf[(size_t)(d0 + 1) * NFFT + k] = Hb;
    }
}

// =====================================================================
// Kernel 3: long conv, TWO channels per FFT, contiguous float2 I/O.
// =====================================================================
__global__ __launch_bounds__(256) void conv_kernel(const float* __restrict__ zn,
                                                   const float2* __restrict__ Wf,
                                                   float* __restrict__ cwhz) {
    const int blk = blockIdx.x;             // 0 .. 4095
    const int b = blk >> 9;
    const int d0 = (blk & 511) * 2;
    __shared__ float2 buf0[NFFT];
    __shared__ float2 buf1[NFFT];
    const float2* src = (const float2*)(zn + ((size_t)b * LEN) * DIM + d0);
    for (int l = threadIdx.x; l < LEN; l += 256)
        buf0[l] = src[(size_t)l * (DIM / 2)];
    for (int l = LEN + threadIdx.x; l < NFFT; l += 256) {
        float2 zv; zv.x = 0.f; zv.y = 0.f;
        buf0[l] = zv;
    }
    float2* Z  = fft8192(buf0, buf1, threadIdx.x, 256, 1.0f);
    float2* Yb = (Z == buf0) ? buf1 : buf0;
    const float2* Ha = Wf + (size_t)d0 * NFFT;
    const float2* Hb = Wf + (size_t)(d0 + 1) * NFFT;
    for (int k = threadIdx.x; k < NFFT; k += 256) {
        const int nk = (NFFT - k) & (NFFT - 1);
        float2 zk = Z[k], zn_ = Z[nk];
        float2 A;  A.x  = 0.5f * (zk.x + zn_.x); A.y  = 0.5f * (zk.y - zn_.y);
        float2 Bv; Bv.x = 0.5f * (zk.y + zn_.y); Bv.y = -0.5f * (zk.x - zn_.x);
        float2 ha = Ha[k], hb = Hb[k];
        float2 P; P.x = A.x * ha.x - A.y * ha.y;  P.y = A.x * ha.y + A.y * ha.x;
        float2 Q; Q.x = Bv.x * hb.x - Bv.y * hb.y; Q.y = Bv.x * hb.y + Bv.y * hb.x;
        float2 Y; Y.x = P.x - Q.y; Y.y = P.y + Q.x;   // P + i*Q
        Yb[k] = Y;
    }
    float2* res = fft8192(Yb, Z, threadIdx.x, 256, -1.0f);
    float2* dst = (float2*)(cwhz + ((size_t)b * LEN) * DIM + d0);
    const float inv = 1.0f / (float)NFFT;
    for (int l = threadIdx.x; l < LEN; l += 256) {
        float2 r = res[l];
        float2 o; o.x = r.x * inv; o.y = r.y * inv;
        dst[(size_t)l * (DIM / 2)] = o;
    }
}

// =====================================================================
// helpers for the WMMA kernel
// =====================================================================
__device__ inline v16bf frag_bf16(const __bf16* p) {
    u4 lo = *(const u4*)(p);
    u4 hi = *(const u4*)(p + 16);
    v16bf r;
    __builtin_memcpy(&r, &lo, 16);
    __builtin_memcpy(((char*)&r) + 16, &hi, 16);
    return r;
}

// Launder only the 32-bit element offset: keeps the __shared__ base (so
// infer-address-space still emits ds_load_b128, not flat_load), while the
// opaque per-iteration value stops LICM from caching all A-fragments in
// VGPRs (which spilled to scratch two rounds ago).
__device__ inline int launder_idx(int u) {
    asm volatile("" : "+v"(u));
    return u;
}

__device__ inline float gelu_exact(float v) {
    return 0.5f * v * (1.0f + erff(v * 0.70710678118654752f));
}

#define WMMA_BF16(A, B, C) \
    __builtin_amdgcn_wmma_f32_16x16x32_bf16(false, (A), false, (B), (short)0, (C), false, false)

// =====================================================================
// Kernel 4: fused gated-MLP + combine, 32 token rows per block.
//  - X tile (32x1024 bf16 = 64 KB) async-staged to LDS
//    (global_load_async_to_lds_b128 + s_wait_asynccnt).
//  - 2x2 register tiles: 4 accumulators, 2 A-frags (LDS, ds_load_b128)
//    + 2 B-frags (L2-resident bf16 weights) -> 4 wmma per 4 loads.
//  - GELU'd 32x2048 intermediate in LDS (128.5 KB). Total LDS ~195 KB
//    of the 320 KB WGP pool.
// =====================================================================
__global__ __launch_bounds__(512) void mlp_kernel(const __bf16* __restrict__ xb,
                                                  const __bf16* __restrict__ W1b,
                                                  const __bf16* __restrict__ W2b,
                                                  const float* __restrict__ cwhz,
                                                  float* __restrict__ out) {
    __shared__ __align__(16) __bf16 Xs[MTILE * XS_LD];   // ~66 KB
    __shared__ __align__(16) __bf16 Ys[MTILE * YS_LD];   // ~128.5 KB
    const int m0    = blockIdx.x * MTILE;
    const int lane  = threadIdx.x & 31;
    const int wave  = threadIdx.x >> 5;
    const int idx16 = lane & 15;
    const int half  = lane >> 4;

    // ---- async stage x tile: 32 rows x 128 chunks of 16 B ----
    {
        const char* gsrc = (const char*)(xb + (size_t)m0 * DIM);
#if defined(__AMDGCN__)
        const unsigned ldsbase = (unsigned)(size_t)(&Xs[0]);
        for (int c = threadIdx.x; c < MTILE * 128; c += 512) {
            const int row = c >> 7, col = c & 127;
            unsigned loff = ldsbase + (unsigned)(row * (XS_LD * 2) + col * 16);
            unsigned long long ga =
                (unsigned long long)(size_t)(gsrc + (size_t)row * (DIM * 2) + col * 16);
            asm volatile("global_load_async_to_lds_b128 %0, %1, off"
                         :: "v"(loff), "v"(ga) : "memory");
        }
        asm volatile("s_wait_asynccnt 0x0" ::: "memory");
#else
        for (int c = threadIdx.x; c < MTILE * 128; c += 512) {
            const int row = c >> 7, col = c & 127;
            ((u4*)(Xs + row * XS_LD))[col] =
                ((const u4*)(gsrc + (size_t)row * (DIM * 2)))[col];
        }
#endif
    }
    __syncthreads();

    // ---------------- GEMM1 + GELU : Y[32,2048] ----------------
    // wave owns n-tiles [8w, 8w+8), both 16-row m-subtiles
    #pragma unroll 1
    for (int j = 0; j < 8; j += 2) {
        const int n0 = (8 * wave + j) * 16;
        const int n1 = n0 + 16;
        v8f acc00 = {}, acc01 = {}, acc10 = {}, acc11 = {};
        const int oA0 = launder_idx((     idx16) * XS_LD + half * 8);
        const int oA1 = launder_idx((16 + idx16) * XS_LD + half * 8);
        const __bf16* pB0 = W1b + (size_t)(n0 + idx16) * DIM + half * 8;
        const __bf16* pB1 = W1b + (size_t)(n1 + idx16) * DIM + half * 8;
        #pragma unroll 2
        for (int k0 = 0; k0 < DIM; k0 += 32) {
            __builtin_prefetch(pB0 + k0 + 128, 0, 3);
            v16bf a0 = frag_bf16(Xs + oA0 + k0);
            v16bf a1 = frag_bf16(Xs + oA1 + k0);
            v16bf b0 = frag_bf16(pB0 + k0);
            v16bf b1 = frag_bf16(pB1 + k0);
            acc00 = WMMA_BF16(a0, b0, acc00);
            acc01 = WMMA_BF16(a0, b1, acc01);
            acc10 = WMMA_BF16(a1, b0, acc10);
            acc11 = WMMA_BF16(a1, b1, acc11);
        }
        #pragma unroll
        for (int r = 0; r < 8; ++r) {
            const int mA = r + half * 8;
            const int mB = 16 + mA;
            Ys[mA * YS_LD + n0 + idx16] = (__bf16)gelu_exact(acc00[r]);
            Ys[mA * YS_LD + n1 + idx16] = (__bf16)gelu_exact(acc01[r]);
            Ys[mB * YS_LD + n0 + idx16] = (__bf16)gelu_exact(acc10[r]);
            Ys[mB * YS_LD + n1 + idx16] = (__bf16)gelu_exact(acc11[r]);
        }
    }
    __syncthreads();

    // ---------------- GEMM2 + combine : G[32,1024] ----------------
    // wave owns n-tiles [4w, 4w+4), both m-subtiles
    #pragma unroll 1
    for (int j = 0; j < 4; j += 2) {
        const int n0 = (4 * wave + j) * 16;
        const int n1 = n0 + 16;
        v8f acc00 = {}, acc01 = {}, acc10 = {}, acc11 = {};
        const int oA0 = launder_idx((     idx16) * YS_LD + half * 8);
        const int oA1 = launder_idx((16 + idx16) * YS_LD + half * 8);
        const __bf16* pB0 = W2b + (size_t)(n0 + idx16) * HID + half * 8;
        const __bf16* pB1 = W2b + (size_t)(n1 + idx16) * HID + half * 8;
        #pragma unroll 2
        for (int k0 = 0; k0 < HID; k0 += 32) {
            __builtin_prefetch(pB0 + k0 + 128, 0, 3);
            v16bf a0 = frag_bf16(Ys + oA0 + k0);
            v16bf a1 = frag_bf16(Ys + oA1 + k0);
            v16bf b0 = frag_bf16(pB0 + k0);
            v16bf b1 = frag_bf16(pB1 + k0);
            acc00 = WMMA_BF16(a0, b0, acc00);
            acc01 = WMMA_BF16(a0, b1, acc01);
            acc10 = WMMA_BF16(a1, b0, acc10);
            acc11 = WMMA_BF16(a1, b1, acc11);
        }
        #pragma unroll
        for (int r = 0; r < 8; ++r) {
            const int mA = r + half * 8;
            const int mB = 16 + mA;
            const size_t iA0 = (size_t)(m0 + mA) * DIM + n0 + idx16;
            const size_t iA1 = (size_t)(m0 + mA) * DIM + n1 + idx16;
            const size_t iB0 = (size_t)(m0 + mB) * DIM + n0 + idx16;
            const size_t iB1 = (size_t)(m0 + mB) * DIM + n1 + idx16;
            out[iA0] = acc00[r] * cwhz[iA0] + out[iA0];   // out held zn
            out[iA1] = acc01[r] * cwhz[iA1] + out[iA1];
            out[iB0] = acc10[r] * cwhz[iB0] + out[iB0];
            out[iB1] = acc11[r] * cwhz[iB1] + out[iB1];
        }
    }
}

// =====================================================================
// launch
// =====================================================================
extern "C" void kernel_launch(void* const* d_in, const int* in_sizes, int n_in,
                              void* d_out, int out_size, void* d_ws, size_t ws_size,
                              hipStream_t stream) {
    const float* z     = (const float*)d_in[0];
    const float* x     = (const float*)d_in[1];
    const float* a     = (const float*)d_in[2];
    const float* pe    = (const float*)d_in[3];
    const float* W_pos = (const float*)d_in[4];
    const float* W1    = (const float*)d_in[5];
    const float* W2    = (const float*)d_in[6];
    const float* gamma = (const float*)d_in[7];
    const float* beta  = (const float*)d_in[8];
    float* out = (float*)d_out;

    // workspace layout (bytes):
    //   [0,        64M)  Wf   : 1024 x 8192 complex spectra
    //   [64M,     192M)  cwhz : 32768 x 1024 f32
    //   [192M,    256M)  xbf  : 32768 x 1024 bf16
    //   [256M,    260M)  W1bf : 2048 x 1024 bf16
    //   [260M,    264M)  W2bf : 1024 x 2048 bf16
    char* ws = (char*)d_ws;
    float2* Wf   = (float2*)(ws);
    float*  cwhz = (float*) (ws + (size_t)DIM * NFFT * sizeof(float2));
    __bf16* xbf  = (__bf16*)(ws + (size_t)192 * 1024 * 1024);
    __bf16* W1bf = (__bf16*)(ws + (size_t)256 * 1024 * 1024);
    __bf16* W2bf = (__bf16*)(ws + (size_t)260 * 1024 * 1024);

    const long nX = (long)NROWS * DIM;        // 33.5M
    const long nW = (long)HID * DIM;          // 2.1M

    f32_to_bf16_kernel<<<(int)(nX / (256 * 8)), 256, 0, stream>>>(x,  xbf,  nX);
    f32_to_bf16_kernel<<<(int)(nW / (256 * 8)), 256, 0, stream>>>(W1, W1bf, nW);
    f32_to_bf16_kernel<<<(int)(nW / (256 * 8)), 256, 0, stream>>>(W2, W2bf, nW);

    ln_kernel        <<<NROWS,           256, 0, stream>>>(z, gamma, beta, out);
    filter_fft_kernel<<<DIM / 2,         256, 0, stream>>>(a, pe, W_pos, Wf);
    conv_kernel      <<<BATCH * DIM / 2, 256, 0, stream>>>(out, Wf, cwhz);
    mlp_kernel       <<<NROWS / MTILE,   512, 0, stream>>>(xbf, W1bf, W2bf, cwhz, out);
}
